// Attention_65420941853381
// MI455X (gfx1250) — compile-verified, matrix operand order checked by reading, not run on hardware
//
#include <hip/hip_runtime.h>
#include <math.h>

// Problem constants (match reference)
#define BB 2
#define LL 2048
#define DD 1024
#define HH 16
#define DH 64
#define NTOK (BB * LL)   // 4096 rows
#define NEGBIG 1.0e7f

typedef __attribute__((ext_vector_type(16))) _Float16 v16h;
typedef __attribute__((ext_vector_type(8)))  _Float16 v8h;
typedef __attribute__((ext_vector_type(8)))  float    v8f;
typedef __attribute__((ext_vector_type(4)))  unsigned int u32x4;
typedef __attribute__((ext_vector_type(8)))  int      i32x8;
typedef __attribute__((ext_vector_type(4)))  int      i32x4;

#if __has_builtin(__builtin_amdgcn_tensor_load_to_lds) && __has_builtin(__builtin_amdgcn_s_wait_tensorcnt)
#define HAVE_TDM 1
#else
#define HAVE_TDM 0
#endif

// ---------------------------------------------------------------------------
// Fragment loader for V_WMMA_F32_16X16X32_F16 operands (wave32).
// 16-bit A-matrix 16x32 layout (ISA 7.12.2): lane L holds row M=L%16;
// halves 0..7  = K =  (L<16 ? 0 : 8) .. +7
// halves 8..15 = K = 16 + (L<16 ? 0 : 8) .. +7
// B-matrix 32x16 is symmetric with lane = column N, K contiguous,
// i.e. load from row-major [N][K] with the identical addressing.
// ---------------------------------------------------------------------------
static __device__ __forceinline__ v16h load_frag(const _Float16* base, int stride, int lane) {
    int row  = lane & 15;
    int koff = (lane >> 4) << 3;            // 0 or 8 halves
    const _Float16* p = base + row * stride + koff;
    v8h lo = *(const v8h*)(p);              // 16B -> b128 load
    v8h hi = *(const v8h*)(p + 16);
    v16h r;
#pragma unroll
    for (int i = 0; i < 8; ++i) { r[i] = lo[i]; r[i + 8] = hi[i]; }
    return r;
}

static __device__ __forceinline__ v8f wmma16(v16h a, v16h b, v8f c) {
    // (neg_a, A, neg_b, B, c_mod, C, reuse_a, reuse_b)
    return __builtin_amdgcn_wmma_f32_16x16x32_f16(false, a, false, b, (short)0, c, false, false);
}

#if HAVE_TDM
// ---------------------------------------------------------------------------
// Tensor Data Mover: async-load a 2D tile (rows x cols f16 elements, global
// row stride in elements) into LDS at byte offset lds_off, with hardware
// LDS padding so rows land on a 72-half (144 B) stride:
//   pad_interval code 4 -> pad after 32 DWORDs (128 B = 64 halves)
//   pad_amount   code 3 -> insert 4 DWORDs (16 B = 8 halves)
// D# bit layout per CDNA5 ISA ch.8 (group0: count/lds/global/type,
// group1: data_size=2B, pad, tensor dims, tile dims, dim0 stride).
// This toolchain exposes the 6-arg builtin:
//   (u32x4 g0, i32x8 g1, i32x4 g2, i32x4 g3, i32x8 extra, i32 cpol)
// ---------------------------------------------------------------------------
static __device__ __forceinline__ void tdm_load_tile_2d(unsigned lds_off,
                                                        const _Float16* gptr,
                                                        unsigned rows,
                                                        unsigned cols,
                                                        unsigned row_stride_elems) {
    unsigned long long ga = (unsigned long long)(size_t)gptr;
    u32x4 g0;
    g0[0] = 1u;                                            // count=1, user mode
    g0[1] = lds_off;                                       // lds_addr (bytes)
    g0[2] = (unsigned)(ga & 0xFFFFFFFFu);                  // global_addr[31:0]
    g0[3] = (unsigned)((ga >> 32) & 0x01FFFFFFu) | (2u << 30); // addr[56:32], type=2
    i32x8 g1;
    g1[0] = (int)((1u << 16) |                             // data_size = 2 bytes
                  (1u << 20) |                             // pad_enable
                  (4u << 22) |                             // pad_interval: 32 DWORDs
                  (3u << 25));                             // pad_amount:   4 DWORDs
    g1[1] = (int)((cols & 0xFFFFu) << 16);                 // tensor_dim0[15:0]
    g1[2] = (int)(((cols >> 16) & 0xFFFFu) | ((rows & 0xFFFFu) << 16)); // dim0 hi, dim1 lo
    g1[3] = (int)(((rows >> 16) & 0xFFFFu) | ((cols & 0xFFFFu) << 16)); // dim1 hi, tile_dim0
    g1[4] = (int)(rows & 0xFFFFu);                         // tile_dim1, tile_dim2=0
    g1[5] = (int)row_stride_elems;                         // tensor_dim0_stride[31:0]
    g1[6] = 0;                                             // stride hi, dim1_stride lo
    g1[7] = 0;
    i32x4 z4 = {0, 0, 0, 0};
    i32x8 z8 = {0, 0, 0, 0, 0, 0, 0, 0};
    __builtin_amdgcn_tensor_load_to_lds(g0, g1, z4, z4, z8, 0);
}
#endif

// ---------------------------------------------------------------------------
// Prep: X_q = f16(q + pe), X_k = f16(k + pe), X_v = f16(v)
// Interleaved sinusoidal PE: even col sin, odd col cos, period 10000^(-i/512)
// ---------------------------------------------------------------------------
__global__ void prep_inputs_kernel(const float* __restrict__ q,
                                   const float* __restrict__ k,
                                   const float* __restrict__ v,
                                   _Float16* __restrict__ Xq,
                                   _Float16* __restrict__ Xk,
                                   _Float16* __restrict__ Xv) {
    int idx = blockIdx.x * blockDim.x + threadIdx.x;
    if (idx >= NTOK * DD) return;
    int dm = idx & (DD - 1);
    int l  = (idx >> 10) & (LL - 1);        // DD = 1024 = 2^10
    int hi = dm >> 1;
    float period = __powf(10000.0f, -(float)hi * (1.0f / (DD / 2)));
    float ang = (float)l * period;
    float pe = (dm & 1) ? __cosf(ang) : __sinf(ang);
    Xq[idx] = (_Float16)(q[idx] + pe);
    Xk[idx] = (_Float16)(k[idx] + pe);
    Xv[idx] = (_Float16)(v[idx]);
}

__global__ void cvt_f32_to_f16_kernel(const float* __restrict__ in,
                                      _Float16* __restrict__ out, int n) {
    int idx = blockIdx.x * blockDim.x + threadIdx.x;
    if (idx < n) out[idx] = (_Float16)in[idx];
}

// ---------------------------------------------------------------------------
// WMMA GEMM:  Y[M,N] = X[M,K] * W[N,K]^T + bias   (torch Linear convention)
// block = 128 threads = 4 waves; each block owns a 64x64 tile of Y;
// each wave owns a 16x64 strip. K-loop step 32 -> 4 WMMAs / wave / step.
// ---------------------------------------------------------------------------
template <bool HALF_OUT>
__global__ void __launch_bounds__(128)
gemm_wmma_kernel(const _Float16* __restrict__ X, const _Float16* __restrict__ W,
                 const float* __restrict__ bias, void* __restrict__ Yv,
                 int M, int N, int K) {
    int lane = threadIdx.x & 31;
    int wave = threadIdx.x >> 5;
    int m0 = blockIdx.x * 64 + wave * 16;
    int n0 = blockIdx.y * 64;

    v8f acc[4] = {};
    const _Float16* Xa = X + (size_t)m0 * K;
    const _Float16* Wb = W + (size_t)n0 * K;

    for (int k = 0; k < K; k += 32) {
        __builtin_prefetch(Xa + k + 128, 0, 0);     // global_prefetch_b8
        v16h a = load_frag(Xa + k, K, lane);
#pragma unroll
        for (int nt = 0; nt < 4; ++nt) {
            v16h b = load_frag(Wb + (size_t)(nt * 16) * K + k, K, lane);
            acc[nt] = wmma16(a, b, acc[nt]);
        }
    }

    // C/D layout: VGPR r -> M = r + (lane>=16 ? 8 : 0); N = lane%16
    int n = lane & 15;
    int mb = (lane >> 4) << 3;
#pragma unroll
    for (int nt = 0; nt < 4; ++nt) {
#pragma unroll
        for (int r = 0; r < 8; ++r) {
            int row = m0 + mb + r;
            int col = n0 + nt * 16 + n;
            float val = acc[nt][r] + bias[col];
            if (HALF_OUT) ((_Float16*)Yv)[(size_t)row * N + col] = (_Float16)val;
            else          ((float*)Yv)[(size_t)row * N + col] = val;
        }
    }
}

// ---------------------------------------------------------------------------
// Flash attention per (b, h, 64-query tile). block = 128 threads = 4 waves.
// Each wave owns 16 query rows. K tile staged via the Tensor Data Mover
// (hardware-padded to the 72-half LDS stride); V tile transposed on fill by
// the 4 waves in parallel with the TDM transfer. Online softmax with running
// row max/sum; causal + key-padding mask. Output f16 in b l (h d) layout.
// ---------------------------------------------------------------------------
__global__ void __launch_bounds__(128)
attn_wmma_kernel(const _Float16* __restrict__ Qp, const _Float16* __restrict__ Kp,
                 const _Float16* __restrict__ Vp, const unsigned char* __restrict__ mask,
                 _Float16* __restrict__ AttO) {
    __shared__ __align__(16) _Float16 Ks[64 * 72];   // [key][d]
    __shared__ __align__(16) _Float16 Vs[64 * 72];   // [d][key]  (transposed)
    __shared__ __align__(16) _Float16 Ps[64 * 72];   // [q][key]  probs (f16)
    __shared__ float Sf[64 * 68];                    // [q][key]  scores (f32)
    __shared__ float rowmax[64], rowsum[64], alphav[64];

    int lane = threadIdx.x & 31;
    int wave = threadIdx.x >> 5;
    int qt = blockIdx.x, h = blockIdx.y, b = blockIdx.z;
    int q0 = qt * 64;

    // Q strip for this wave stays resident in registers for the whole loop.
    const _Float16* Qbase = Qp + ((size_t)(b * LL + q0 + wave * 16)) * DD + h * DH;
    v16h aq0 = load_frag(Qbase + 0,  DD, lane);
    v16h aq1 = load_frag(Qbase + 32, DD, lane);

    v8f o[4] = {};
    if (threadIdx.x < 64) { rowmax[threadIdx.x] = -INFINITY; rowsum[threadIdx.x] = 0.0f; }
    __syncthreads();

    for (int kt = 0; kt <= qt; ++kt) {
        int k0 = kt * 64;
        const _Float16* Kt = Kp + ((size_t)(b * LL + k0)) * DD + h * DH;
        const _Float16* Vt = Vp + ((size_t)(b * LL + k0)) * DD + h * DH;

#if HAVE_TDM
        // Wave 0 kicks off the TDM transfer of the K tile (64x64 f16,
        // global row stride DD elements, LDS rows padded to 72 halves).
        if (wave == 0) {
            tdm_load_tile_2d((unsigned)(size_t)(&Ks[0]), Kt, 64u, 64u, (unsigned)DD);
        }
        // All 4 waves transpose-stage the V tile meanwhile.
        for (int i = threadIdx.x; i < 64 * 64; i += 128) {
            int kk = i >> 6;       // key row within tile
            int dd = i & 63;       // d within head
            Vs[dd * 72 + kk] = Vt[(size_t)kk * DD + dd];
        }
        if (wave == 0) __builtin_amdgcn_s_wait_tensorcnt(0);
#else
        for (int i = threadIdx.x; i < 64 * 64; i += 128) {
            int kk = i >> 6;
            int dd = i & 63;
            Ks[kk * 72 + dd] = Kt[(size_t)kk * DD + dd];
            Vs[dd * 72 + kk] = Vt[(size_t)kk * DD + dd];
        }
#endif
        __syncthreads();

        // S = Q * K^T  (scaled + masked on the way to LDS)
        int n = lane & 15;
        int mb = (lane >> 4) << 3;
#pragma unroll
        for (int nt = 0; nt < 4; ++nt) {
            v16h bk0 = load_frag(&Ks[(nt * 16) * 72 + 0],  72, lane);
            v16h bk1 = load_frag(&Ks[(nt * 16) * 72 + 32], 72, lane);
            v8f s = {};
            s = wmma16(aq0, bk0, s);
            s = wmma16(aq1, bk1, s);
#pragma unroll
            for (int r = 0; r < 8; ++r) {
                int ml = wave * 16 + mb + r;         // local query row
                int qi = q0 + ml;
                int kj = k0 + nt * 16 + n;
                float val = s[r] * 0.125f;           // 1/sqrt(64)
                if ((kj > qi) || mask[b * LL + kj]) val -= NEGBIG;
                Sf[ml * 68 + nt * 16 + n] = val;
            }
        }
        __syncthreads();

        // Online softmax: one thread per query row.
        if (threadIdx.x < 64) {
            int rr = threadIdx.x;
            float om = rowmax[rr];
            float tm = om;
            for (int c = 0; c < 64; ++c) tm = fmaxf(tm, Sf[rr * 68 + c]);
            float al = __expf(om - tm);
            float sum = rowsum[rr] * al;
            for (int c = 0; c < 64; ++c) {
                float p = __expf(Sf[rr * 68 + c] - tm);
                sum += p;
                Ps[rr * 72 + c] = (_Float16)p;
            }
            rowmax[rr] = tm; rowsum[rr] = sum; alphav[rr] = al;
        }
        __syncthreads();

        // Rescale running O by alpha, then O += P * V.
        v16h ap0 = load_frag(&Ps[(wave * 16) * 72 + 0],  72, lane);
        v16h ap1 = load_frag(&Ps[(wave * 16) * 72 + 32], 72, lane);
#pragma unroll
        for (int nt = 0; nt < 4; ++nt) {
#pragma unroll
            for (int r = 0; r < 8; ++r) o[nt][r] *= alphav[wave * 16 + mb + r];
            v16h bv0 = load_frag(&Vs[(nt * 16) * 72 + 0],  72, lane);
            v16h bv1 = load_frag(&Vs[(nt * 16) * 72 + 32], 72, lane);
            o[nt] = wmma16(ap0, bv0, o[nt]);
            o[nt] = wmma16(ap1, bv1, o[nt]);
        }
        __syncthreads();
    }

    // Normalize and store as f16 in b l (h d) layout.
    int n = lane & 15;
    int mb = (lane >> 4) << 3;
#pragma unroll
    for (int nt = 0; nt < 4; ++nt) {
#pragma unroll
        for (int r = 0; r < 8; ++r) {
            int ml = wave * 16 + mb + r;
            int qi = q0 + ml;
            float recip = 1.0f / rowsum[ml];
            AttO[((size_t)(b * LL + qi)) * DD + h * DH + nt * 16 + n] =
                (_Float16)(o[nt][r] * recip);
        }
    }
}

// ---------------------------------------------------------------------------
// Host-side orchestration. Inputs (dict order):
// 0:q 1:k 2:v 3:mask 4:Wq 5:bq 6:Wk 7:bk 8:Wv 9:bv 10:Wo 11:bo
// ---------------------------------------------------------------------------
extern "C" void kernel_launch(void* const* d_in, const int* in_sizes, int n_in,
                              void* d_out, int out_size, void* d_ws, size_t ws_size,
                              hipStream_t stream) {
    (void)in_sizes; (void)n_in; (void)out_size; (void)ws_size;

    const float* q  = (const float*)d_in[0];
    const float* k  = (const float*)d_in[1];
    const float* v  = (const float*)d_in[2];
    const unsigned char* mask = (const unsigned char*)d_in[3];
    const float* Wq = (const float*)d_in[4];
    const float* bq = (const float*)d_in[5];
    const float* Wk = (const float*)d_in[6];
    const float* bk = (const float*)d_in[7];
    const float* Wv = (const float*)d_in[8];
    const float* bv = (const float*)d_in[9];
    const float* Wo = (const float*)d_in[10];
    const float* bo = (const float*)d_in[11];
    float* out = (float*)d_out;

    // Workspace layout (f16), total 32M halves = 64 MB.
    _Float16* Xq   = (_Float16*)d_ws;
    _Float16* Xk   = Xq  + (size_t)NTOK * DD;
    _Float16* Xv   = Xk  + (size_t)NTOK * DD;
    _Float16* Wqh  = Xv  + (size_t)NTOK * DD;
    _Float16* Wkh  = Wqh + (size_t)DD * DD;
    _Float16* Wvh  = Wkh + (size_t)DD * DD;
    _Float16* Woh  = Wvh + (size_t)DD * DD;
    _Float16* Qp   = Woh + (size_t)DD * DD;
    _Float16* Kp   = Qp  + (size_t)NTOK * DD;
    _Float16* Vp   = Kp  + (size_t)NTOK * DD;
    _Float16* AttO = Vp  + (size_t)NTOK * DD;

    // 1) PE add + f16 conversion of activations.
    {
        int total = NTOK * DD;
        prep_inputs_kernel<<<(total + 255) / 256, 256, 0, stream>>>(q, k, v, Xq, Xk, Xv);
    }
    // 2) Weights to f16 (row-major W is already the WMMA B operand for x@W.T).
    {
        int nw = DD * DD;
        int g = (nw + 255) / 256;
        cvt_f32_to_f16_kernel<<<g, 256, 0, stream>>>(Wq, Wqh, nw);
        cvt_f32_to_f16_kernel<<<g, 256, 0, stream>>>(Wk, Wkh, nw);
        cvt_f32_to_f16_kernel<<<g, 256, 0, stream>>>(Wv, Wvh, nw);
        cvt_f32_to_f16_kernel<<<g, 256, 0, stream>>>(Wo, Woh, nw);
    }
    // 3) Fused QKV projections as three WMMA GEMMs (f16 out).
    {
        dim3 grid(NTOK / 64, DD / 64);
        gemm_wmma_kernel<true><<<grid, 128, 0, stream>>>(Xq, Wqh, bq, Qp, NTOK, DD, DD);
        gemm_wmma_kernel<true><<<grid, 128, 0, stream>>>(Xk, Wkh, bk, Kp, NTOK, DD, DD);
        gemm_wmma_kernel<true><<<grid, 128, 0, stream>>>(Xv, Wvh, bv, Vp, NTOK, DD, DD);
    }
    // 4) Causal + key-padding flash attention (WMMA core, TDM staging).
    {
        dim3 grid(LL / 64, HH, BB);
        attn_wmma_kernel<<<grid, 128, 0, stream>>>(Qp, Kp, Vp, mask, AttO);
    }
    // 5) Output projection straight into d_out (f32).
    {
        dim3 grid(NTOK / 64, DD / 64);
        gemm_wmma_kernel<false><<<grid, 128, 0, stream>>>(AttO, Woh, bo, out, NTOK, DD, DD);
    }
}
